// APINNBeam_53386443489714
// MI455X (gfx1250) — matrix operand order repaired and checked by reading, not compiled, not run on hardware
//
#include <hip/hip_runtime.h>
#include <hip/hip_bf16.h>

typedef __attribute__((ext_vector_type(16))) _Float16 v16h;
typedef __attribute__((ext_vector_type(8)))  _Float16 v8h;
typedef __attribute__((ext_vector_type(8)))  float    v8f;

#define WAVES_PER_BLOCK 4
#define THREADS (WAVES_PER_BLOCK * 32)
#define ACT_STRIDE 128   // halves per activation row (112 padded to 128)

// ---------------- fast transcendental helpers (TRANS unit, co-executes with WMMA) -------------
__device__ __forceinline__ float fast_tanh(float x) {
#if __has_builtin(__builtin_amdgcn_tanhf)
    return __builtin_amdgcn_tanhf(x);           // native v_tanh_f32 (CDNA5 TRANS op)
#else
    float t = x * 2.8853900817779268f;          // 2*log2(e)
    t = fminf(fmaxf(t, -30.f), 30.f);
    float e = __builtin_amdgcn_exp2f(t);        // e^(2x)
    return (e - 1.f) * __builtin_amdgcn_rcpf(e + 1.f);
#endif
}
__device__ __forceinline__ float fast_softplus(float x) {
    float e  = __builtin_amdgcn_exp2f(fminf(x, 15.f) * 1.4426950408889634f);
    // v_log_f32 computes log2; __builtin_amdgcn_logf is the log2 builtin on AMDGPU
    float sp = __builtin_amdgcn_logf(1.f + e) * 0.6931471805599453f;
    return (x > 15.f) ? x : sp;
}

// ---------------- weight staging: global f32 -> LDS f16, zero padded ----------------
template<int DIN, int DOUT>
__device__ __forceinline__ void stage(const float* __restrict__ W, const float* __restrict__ B,
                                      _Float16* __restrict__ Wl, float* __restrict__ Bl, int tid) {
    constexpr int KPAD = ((DIN  + 31) / 32) * 32;
    constexpr int NPAD = ((DOUT + 15) / 16) * 16;
    __syncthreads();                       // prior readers of Wl/Bl done
    for (int i = tid; i < KPAD * NPAD; i += THREADS) {
        int k = i / NPAD, n = i - k * NPAD;
        float v = (k < DIN && n < DOUT) ? W[k * DOUT + n] : 0.f;
        Wl[n * KPAD + k] = (_Float16)v;
    }
    for (int i = tid; i < NPAD; i += THREADS)
        Bl[i] = (i < DOUT) ? B[i] : 0.f;
    __syncthreads();                       // staged data visible
}

union AF { v16h v; v8h h[2]; };

// ---------------- generic WMMA layer: LDS act[16][128] f16 -> LDS act f16 ----------------
template<int DIN, int DOUT, bool TANH>
__device__ __forceinline__ void wmma_layer(const _Float16* __restrict__ actIn,
                                           _Float16* __restrict__ actOut,
                                           const _Float16* __restrict__ Wl,
                                           const float* __restrict__ Bl, int lane) {
    constexpr int KC   = (DIN + 31) / 32;          // K chunks of 32
    constexpr int NC   = (DOUT + 15) / 16;         // N chunks of 16
    constexpr int NPAD = NC * 16;
    const int m  = lane & 15;
    const int hi = lane >> 4;

    // A fragments (ISA 16-bit A 16x32 layout): lane holds K = hi*8..+7 and 16+hi*8..+7
    AF afr[KC];
#pragma unroll
    for (int kb = 0; kb < KC; ++kb) {
        const _Float16* p = actIn + m * ACT_STRIDE + kb * 32 + hi * 8;
        afr[kb].h[0] = *(const v8h*)p;
        afr[kb].h[1] = *(const v8h*)(p + 16);
    }
#pragma unroll
    for (int nb = 0; nb < NC; ++nb) {
        v8f acc = {};
#pragma unroll
        for (int kb = 0; kb < KC; ++kb) {
            // B layout: lane holds col nb*16 + (lane&15), K = kb*32 + hi*8..+7 and 16+hi*8..+7
            const _Float16* q = Wl + (nb * 16 + m) * (KC * 32) + kb * 32 + hi * 8;
            AF bf;
            bf.h[0] = *(const v8h*)q;
            bf.h[1] = *(const v8h*)(q + 16);
            acc = __builtin_amdgcn_wmma_f32_16x16x32_f16(false, afr[kb].v, false, bf.v,
                                                         (short)0, acc, false, false);
        }
        const float bias = Bl[nb * 16 + m];
#pragma unroll
        for (int r = 0; r < 8; ++r) {           // D: lane holds col (lane&15), rows r+8*hi
            float v = acc[r] + bias;
            if (TANH) v = fast_tanh(v);
            actOut[(r + 8 * hi) * ACT_STRIDE + nb * 16 + m] = (_Float16)v;
        }
    }
    // zero K-pad columns so next layer's padded K chunk contributes 0
    constexpr int KPN = ((DOUT + 31) / 32) * 32;
    if (KPN > DOUT) {                           // pad is exactly 16 cols: DOUT..DOUT+15
#pragma unroll
        for (int r = 0; r < 8; ++r)
            actOut[(r + 8 * hi) * ACT_STRIDE + DOUT + m] = (_Float16)0.f;
    }
}

// ---------------- final WMMA layer (DOUT<=16): f32 result + bias -> small LDS f32 buffer ----
template<int DIN, int DOUT>
__device__ __forceinline__ void wmma_layer_out(const _Float16* __restrict__ actIn,
                                               float* __restrict__ outBuf,
                                               const _Float16* __restrict__ Wl,
                                               const float* __restrict__ Bl, int lane) {
    constexpr int KC = (DIN + 31) / 32;
    constexpr int NPAD = 16;
    const int m = lane & 15, hi = lane >> 4;
    v8f acc = {};
#pragma unroll
    for (int kb = 0; kb < KC; ++kb) {
        const _Float16* p = actIn + m * ACT_STRIDE + kb * 32 + hi * 8;
        AF a; a.h[0] = *(const v8h*)p; a.h[1] = *(const v8h*)(p + 16);
        const _Float16* q = Wl + m * (KC * 32) + kb * 32 + hi * 8;
        AF b; b.h[0] = *(const v8h*)q; b.h[1] = *(const v8h*)(q + 16);
        acc = __builtin_amdgcn_wmma_f32_16x16x32_f16(false, a.v, false, b.v,
                                                     (short)0, acc, false, false);
    }
    if (m < DOUT) {
        const float bias = Bl[m];
#pragma unroll
        for (int r = 0; r < 8; ++r)
            outBuf[(r + 8 * hi) * DOUT + m] = acc[r] + bias;
    }
}

// =====================================================================================
__global__ __launch_bounds__(THREADS) void apinn_beam_kernel(
    const float* __restrict__ x,
    const float* __restrict__ sW0, const float* __restrict__ sB0,
    const float* __restrict__ sW1, const float* __restrict__ sB1,
    const float* __restrict__ sW2, const float* __restrict__ sB2,
    const float* __restrict__ sW3, const float* __restrict__ sB3,
    const float* __restrict__ eW0, const float* __restrict__ eB0,
    const float* __restrict__ eW1, const float* __restrict__ eB1,
    const float* __restrict__ eW2, const float* __restrict__ eB2,
    const float* __restrict__ eW3, const float* __restrict__ eB3,
    const float* __restrict__ gW0, const float* __restrict__ gB0,
    const float* __restrict__ gW1, const float* __restrict__ gB1,
    const float* __restrict__ gW2, const float* __restrict__ gB2,
    const float* __restrict__ gW3, const float* __restrict__ gB3,
    float* __restrict__ out, int N)
{
    __shared__ __align__(16) _Float16 smW[128 * 112];                       // staged layer weights
    __shared__ float                 smBias[112];
    __shared__ __align__(16) _Float16 smA [WAVES_PER_BLOCK][16 * ACT_STRIDE];
    __shared__ __align__(16) _Float16 smB_[WAVES_PER_BLOCK][16 * ACT_STRIDE];
    __shared__ __align__(16) _Float16 smFeat[WAVES_PER_BLOCK][16 * 64];     // shared feature copy
    __shared__ float                 smSmall[WAVES_PER_BLOCK][16 * 4];      // logits / expert raw
    __shared__ float                 smL0[112 * 2 + 16 * 2];                // sW0,sB0,gW0,gB0

    const int tid  = threadIdx.x;
    const int wave = tid >> 5;
    const int lane = tid & 31;
    const int m    = lane & 15;
    const int hi   = lane >> 4;
    const int tile = blockIdx.x * WAVES_PER_BLOCK + wave;
    const int idx  = tile * 16 + m;

    // stage the tiny 1-wide input layers once
    for (int i = tid; i < 112; i += THREADS) { smL0[i] = sW0[i]; smL0[112 + i] = sB0[i]; }
    for (int i = tid; i < 16;  i += THREADS) { smL0[224 + i] = gW0[i]; smL0[240 + i] = gB0[i]; }
    __syncthreads();

    _Float16* actA = smA [wave];
    _Float16* actB = smB_[wave];
    _Float16* feat = smFeat[wave];
    float*    fbuf = smSmall[wave];

    const float xv = (idx < N) ? x[idx] : 0.f;

    // ---- shared MLP layer 0: 1 -> 112 (VALU outer product + tanh) ----
#pragma unroll 8
    for (int j = 0; j < 56; ++j) {
        const int c = hi * 56 + j;
        actA[m * ACT_STRIDE + c] = (_Float16)fast_tanh(xv * smL0[c] + smL0[112 + c]);
    }
    { v8h z = {}; *(v8h*)(actA + m * ACT_STRIDE + 112 + hi * 8) = z; }   // zero K-pad 112..127

    // ---- shared MLP layers 1..3 ----
    stage<112, 112>(sW1, sB1, smW, smBias, tid);
    wmma_layer<112, 112, true >(actA, actB, smW, smBias, lane);
    stage<112, 112>(sW2, sB2, smW, smBias, tid);
    wmma_layer<112, 112, true >(actB, actA, smW, smBias, lane);
    stage<112, 48 >(sW3, sB3, smW, smBias, tid);
    wmma_layer<112, 48,  false>(actA, actB, smW, smBias, lane);

    // save feature (16 x 64 incl. zeroed pad 48..63) for the expert loop
    {
        const int r = lane >> 1, seg = lane & 1;
#pragma unroll
        for (int c = 0; c < 4; ++c)
            *(v8h*)(feat + r * 64 + seg * 32 + c * 8) =
                *(const v8h*)(actB + r * ACT_STRIDE + seg * 32 + c * 8);
    }

    // ---- gate MLP ----
#pragma unroll
    for (int j = 0; j < 8; ++j) {
        const int c = hi * 8 + j;
        actA[m * ACT_STRIDE + c] = (_Float16)fast_tanh(xv * smL0[224 + c] + smL0[240 + c]);
    }
    { v8h z = {}; *(v8h*)(actA + m * ACT_STRIDE + 16 + hi * 8) = z; }    // zero K-pad 16..31
    stage<16, 16>(gW1, gB1, smW, smBias, tid);
    wmma_layer<16, 16, true>(actA, actB, smW, smBias, lane);
    stage<16, 16>(gW2, gB2, smW, smBias, tid);
    wmma_layer<16, 16, true>(actB, actA, smW, smBias, lane);
    stage<16, 4 >(gW3, gB3, smW, smBias, tid);
    wmma_layer_out<16, 4>(actA, fbuf, smW, smBias, lane);

    // softmax over 4 logits (computed redundantly in both half-waves; no divergence)
    float gl[4], gw[4];
#pragma unroll
    for (int i = 0; i < 4; ++i) gl[i] = fbuf[m * 4 + i];
    float mx = fmaxf(fmaxf(gl[0], gl[1]), fmaxf(gl[2], gl[3]));
    float s = 0.f;
#pragma unroll
    for (int i = 0; i < 4; ++i) { gw[i] = __builtin_amdgcn_exp2f((gl[i] - mx) * 1.4426950408889634f); s += gw[i]; }
    const float rs = __builtin_amdgcn_rcpf(s);
#pragma unroll
    for (int i = 0; i < 4; ++i) gw[i] *= rs;
    if (hi == 0 && idx < N) {
#pragma unroll
        for (int i = 0; i < 4; ++i) {
            out[2 * N + idx * 4 + i] = gw[i];   // weights
            out[6 * N + idx * 4 + i] = gl[i];   // gate_logits
        }
    }

    // ---- experts (fully unrolled so gw[e] stays in registers) ----
    float uacc = 0.f, eiacc = 0.f;
    const float xq = xv * (xv - 1.0f) * 0.01f;    // L = 1.0
#pragma unroll
    for (int e = 0; e < 4; ++e) {
        {   // restore feature into actA (cols 0..63; 48..63 are zero)
            const int r = lane >> 1, seg = lane & 1;
#pragma unroll
            for (int c = 0; c < 4; ++c)
                *(v8h*)(actA + r * ACT_STRIDE + seg * 32 + c * 8) =
                    *(const v8h*)(feat + r * 64 + seg * 32 + c * 8);
        }
        stage<48, 112>(eW0 + e * 48 * 112, eB0 + e * 112, smW, smBias, tid);
        wmma_layer<48, 112, true>(actA, actB, smW, smBias, lane);
        stage<112, 112>(eW1 + e * 112 * 112, eB1 + e * 112, smW, smBias, tid);
        wmma_layer<112, 112, true>(actB, actA, smW, smBias, lane);
        stage<112, 112>(eW2 + e * 112 * 112, eB2 + e * 112, smW, smBias, tid);
        wmma_layer<112, 112, true>(actA, actB, smW, smBias, lane);
        stage<112, 2>(eW3 + e * 112 * 2, eB3 + e * 2, smW, smBias, tid);
        wmma_layer_out<112, 2>(actB, fbuf, smW, smBias, lane);

        const float r0 = fbuf[m * 2 + 0];
        const float r1 = fbuf[m * 2 + 1];
        uacc  += gw[e] * (r0 * xq);
        eiacc += gw[e] * fast_softplus(r1);
    }
    if (hi == 0 && idx < N) {
        out[idx]     = uacc;   // u
        out[N + idx] = eiacc;  // EI
    }
}

extern "C" void kernel_launch(void* const* d_in, const int* in_sizes, int n_in,
                              void* d_out, int out_size, void* d_ws, size_t ws_size,
                              hipStream_t stream) {
    (void)n_in; (void)d_ws; (void)ws_size; (void)out_size;
    const float* x   = (const float*)d_in[0];
    // shared_params: (W,b) x 4 layers
    const float* sW0 = (const float*)d_in[1];  const float* sB0 = (const float*)d_in[2];
    const float* sW1 = (const float*)d_in[3];  const float* sB1 = (const float*)d_in[4];
    const float* sW2 = (const float*)d_in[5];  const float* sB2 = (const float*)d_in[6];
    const float* sW3 = (const float*)d_in[7];  const float* sB3 = (const float*)d_in[8];
    // expert_params: stacked (W[E,din,dout], b[E,dout]) x 4 layers
    const float* eW0 = (const float*)d_in[9];  const float* eB0 = (const float*)d_in[10];
    const float* eW1 = (const float*)d_in[11]; const float* eB1 = (const float*)d_in[12];
    const float* eW2 = (const float*)d_in[13]; const float* eB2 = (const float*)d_in[14];
    const float* eW3 = (const float*)d_in[15]; const float* eB3 = (const float*)d_in[16];
    // gate_params: (W,b) x 4 layers
    const float* gW0 = (const float*)d_in[17]; const float* gB0 = (const float*)d_in[18];
    const float* gW1 = (const float*)d_in[19]; const float* gB1 = (const float*)d_in[20];
    const float* gW2 = (const float*)d_in[21]; const float* gB2 = (const float*)d_in[22];
    const float* gW3 = (const float*)d_in[23]; const float* gB3 = (const float*)d_in[24];

    const int N = in_sizes[0];                     // x is [N,1]
    const int tilesPerBlock = 16 * WAVES_PER_BLOCK;
    const int blocks = (N + tilesPerBlock - 1) / tilesPerBlock;

    apinn_beam_kernel<<<blocks, THREADS, 0, stream>>>(
        x,
        sW0, sB0, sW1, sB1, sW2, sB2, sW3, sB3,
        eW0, eB0, eW1, eB1, eW2, eB2, eW3, eB3,
        gW0, gB0, gW1, gB1, gW2, gB2, gW3, gB3,
        (float*)d_out, N);
}
